// LSTMDecoder_31507880084185
// MI455X (gfx1250) — matrix operand order repaired and checked by reading, not compiled, not run on hardware
//
#include <hip/hip_runtime.h>
#include <hip/hip_bf16.h>
#include <math.h>

typedef _Float16 v16h __attribute__((ext_vector_type(16)));
typedef _Float16 v8h  __attribute__((ext_vector_type(8)));
typedef float    v8f  __attribute__((ext_vector_type(8)));

#define B_SZ 256
#define E_SZ 512
#define T_SZ 512
#define V_SZ 50257
#define HID  1024
#define G4   4096   /* 4*HID */
#define NCHUNK 4            /* T-chunks for one-pass attention */
#define TCHUNK (T_SZ / NCHUNK)

__device__ __forceinline__ float sigmoidf_(float x) { return 1.0f / (1.0f + expf(-x)); }

// ---------------------------------------------------------------------------
// 1) emb = table[z]; x1 = [emb | context]   (256 x 1536, f32)
// ---------------------------------------------------------------------------
__global__ void embed_concat_kernel(const int* __restrict__ z,
                                    const float* __restrict__ ctx,
                                    const float* __restrict__ table,
                                    float* __restrict__ x1) {
    int g = blockIdx.x * blockDim.x + threadIdx.x;   // < 256*1536
    int b = g / (3 * E_SZ);
    int j = g - b * (3 * E_SZ);
    float v;
    if (j < E_SZ) v = table[(size_t)z[b] * E_SZ + j];
    else          v = ctx[(size_t)b * HID + (j - E_SZ)];
    x1[g] = v;
}

// ---------------------------------------------------------------------------
// 2) WMMA GEMM:  C[M=256, N] = A1[256,K1]@W1[K1,N] (+ A2[256,K2]@W2[K2,N]) + bias
//    f32 global data, f16 staged in LDS, v_wmma_f32_16x16x32_f16 compute.
//    Block: 256 threads (8 waves). Tile 64(M) x 128(N), K-step 32.
//    Wave grid 2(m) x 4(n); each wave: 2x2 frags of 16x16.
//    RAGGED=false requires N % 128 == 0 (vectorized, unguarded path).
// ---------------------------------------------------------------------------
template <bool RAGGED>
__launch_bounds__(256)
__global__ void gemm_f16wmma_kernel(const float* __restrict__ A1,
                                    const float* __restrict__ W1, int K1,
                                    const float* __restrict__ A2,
                                    const float* __restrict__ W2, int K2,
                                    const float* __restrict__ bias,
                                    float* __restrict__ C,
                                    int N, int act) {
    __shared__ _Float16 sA[64 * 32];    // [m][k]
    __shared__ _Float16 sB[128 * 32];   // [n][k]  (W tile transposed)

    const int tid  = threadIdx.x;
    const int wave = tid >> 5;
    const int lane = tid & 31;
    const int wm   = wave >> 2;          // 0..1  -> 32 rows each
    const int wn   = wave & 3;           // 0..3  -> 32 cols each
    const int lr   = lane & 15;
    const int hs   = lane >> 4;          // half-select
    const int m0   = blockIdx.y * 64;
    const int n0   = blockIdx.x * 128;

    // A staging coords: 8 floats per thread
    const int ar = tid >> 2;             // 0..63
    const int akc = (tid & 3) << 3;      // 0,8,16,24
    // B staging coords: 2 k-rows x 8 cols per thread, packed half2 along k
    const int kp = tid >> 4;             // 0..15 -> k rows {2kp, 2kp+1}
    const int nc = (tid & 15) << 3;      // 0..120

    const bool fullTile = !RAGGED || (n0 + 128 <= N);   // uniform per block

    v8f acc[2][2];
    for (int i = 0; i < 2; ++i)
        for (int j = 0; j < 2; ++j)
            for (int e = 0; e < 8; ++e) acc[i][j][e] = 0.0f;

    for (int pair = 0; pair < 2; ++pair) {
        const float* A = pair ? A2 : A1;
        const float* W = pair ? W2 : W1;
        const int    K = pair ? K2 : K1;
        if (K <= 0) continue;

        for (int k0 = 0; k0 < K; k0 += 32) {
            // --- stage A tile 64x32 (f32 -> f16), [m][k] ---
            {
                const float* src = A + (size_t)(m0 + ar) * K + k0 + akc;
                float4 f0 = *(const float4*)(src);
                float4 f1 = *(const float4*)(src + 4);
                _Float16* d = &sA[ar * 32 + akc];
                d[0] = (_Float16)f0.x; d[1] = (_Float16)f0.y;
                d[2] = (_Float16)f0.z; d[3] = (_Float16)f0.w;
                d[4] = (_Float16)f1.x; d[5] = (_Float16)f1.y;
                d[6] = (_Float16)f1.z; d[7] = (_Float16)f1.w;
            }
            // --- stage W tile 32x128 transposed -> sB[n][k], half2 packed ---
            {
                const float* r0 = W + (size_t)(k0 + 2 * kp) * N + n0 + nc;
                const float* r1 = r0 + N;
                float v0[8], v1[8];
                if (!RAGGED) {
                    float4 a0 = *(const float4*)(r0);
                    float4 a1 = *(const float4*)(r0 + 4);
                    float4 c0 = *(const float4*)(r1);
                    float4 c1 = *(const float4*)(r1 + 4);
                    v0[0] = a0.x; v0[1] = a0.y; v0[2] = a0.z; v0[3] = a0.w;
                    v0[4] = a1.x; v0[5] = a1.y; v0[6] = a1.z; v0[7] = a1.w;
                    v1[0] = c0.x; v1[1] = c0.y; v1[2] = c0.z; v1[3] = c0.w;
                    v1[4] = c1.x; v1[5] = c1.y; v1[6] = c1.z; v1[7] = c1.w;
                } else if (fullTile) {
                    for (int j = 0; j < 8; ++j) { v0[j] = r0[j]; v1[j] = r1[j]; }
                } else {
                    for (int j = 0; j < 8; ++j) {
                        bool ok = (n0 + nc + j) < N;
                        v0[j] = ok ? r0[j] : 0.0f;
                        v1[j] = ok ? r1[j] : 0.0f;
                    }
                }
                for (int j = 0; j < 8; ++j) {
                    union { _Float16 h[2]; unsigned u; } p;
                    p.h[0] = (_Float16)v0[j];
                    p.h[1] = (_Float16)v1[j];
                    *(unsigned*)(&sB[(nc + j) * 32 + 2 * kp]) = p.u;
                }
            }
            __syncthreads();

            // --- fragment loads (contiguous k = hs*16 + 0..15, same order for A and B) ---
            v16h a[2], b[2];
            for (int mi = 0; mi < 2; ++mi) {
                int row = wm * 32 + mi * 16 + lr;
                union { v16h v; v8h h[2]; } u;
                u.h[0] = *(const v8h*)&sA[row * 32 + hs * 16];       // K = hs*16 + 0..7
                u.h[1] = *(const v8h*)&sA[row * 32 + hs * 16 + 8];   // K = hs*16 + 8..15
                a[mi] = u.v;
            }
            for (int ni = 0; ni < 2; ++ni) {
                int col = wn * 32 + ni * 16 + lr;
                b[ni] = *(const v16h*)&sB[col * 32 + hs * 16];       // K = hs*16 + 0..15
            }

            acc[0][0] = __builtin_amdgcn_wmma_f32_16x16x32_f16(false, a[0], false, b[0], (short)0, acc[0][0], false, false);
            acc[0][1] = __builtin_amdgcn_wmma_f32_16x16x32_f16(false, a[0], false, b[1], (short)0, acc[0][1], false, false);
            acc[1][0] = __builtin_amdgcn_wmma_f32_16x16x32_f16(false, a[1], false, b[0], (short)0, acc[1][0], false, false);
            acc[1][1] = __builtin_amdgcn_wmma_f32_16x16x32_f16(false, a[1], false, b[1], (short)0, acc[1][1], false, false);
            __syncthreads();
        }
    }

    // --- bias + activation + store (D layout: vgpr r -> M = r + hs*8) ---
    for (int mi = 0; mi < 2; ++mi) {
        for (int ni = 0; ni < 2; ++ni) {
            int gc = n0 + wn * 32 + ni * 16 + lr;
            if (gc < N) {
                float bv = bias ? bias[gc] : 0.0f;
                for (int r = 0; r < 8; ++r) {
                    int gr = m0 + wm * 32 + mi * 16 + hs * 8 + r;
                    float v = acc[mi][ni][r] + bv;
                    if (act == 1) v = tanhf(v);
                    C[(size_t)gr * N + gc] = v;
                }
            }
        }
    }
}

// ---------------------------------------------------------------------------
// 3) LSTM cell pointwise:  z (256x4096) + c_prev -> h, c
// ---------------------------------------------------------------------------
__global__ void lstm_cell_kernel(const float* __restrict__ z,
                                 const float* __restrict__ c_prev,
                                 float* __restrict__ h_out,
                                 float* __restrict__ c_out) {
    int idx = blockIdx.x * blockDim.x + threadIdx.x;   // < 256*1024
    int b = idx >> 10, n = idx & (HID - 1);
    const float* zr = z + (size_t)b * G4;
    float ig = sigmoidf_(zr[n]);
    float fg = sigmoidf_(zr[HID + n]);
    float gg = tanhf(zr[2 * HID + n]);
    float og = sigmoidf_(zr[3 * HID + n]);
    float c  = fg * c_prev[idx] + ig * gg;
    float h  = og * tanhf(c);
    c_out[idx] = c;
    h_out[idx] = h;
}

// ---------------------------------------------------------------------------
// 4) One-pass attention over a T-chunk (online softmax, single read of H):
//    grid (NCHUNK, B). Per block: stream H[b, t0..t0+TCHUNK), maintain
//    running max m, running sum l, running weighted ctx accumulator.
//    Writes per-chunk partials; combined by attn_combine_kernel.
// ---------------------------------------------------------------------------
__launch_bounds__(256)
__global__ void attn_fused_kernel(const float* __restrict__ H,
                                  const float* __restrict__ alpha,
                                  float* __restrict__ ctx_part,   // [B*NCHUNK, HID]
                                  float* __restrict__ ml_part) {  // [B*NCHUNK, 2]
    __shared__ float sAlpha[HID];
    __shared__ float red[8];
    const int chunk = blockIdx.x;
    const int b     = blockIdx.y;
    const int tid   = threadIdx.x;
    const int wave  = tid >> 5, lane = tid & 31;

    // load alpha row once
    for (int j = 0; j < 4; ++j) sAlpha[tid + j * 256] = alpha[(size_t)b * HID + tid + j * 256];
    __syncthreads();

    const float* hb = H + ((size_t)b * T_SZ + (size_t)chunk * TCHUNK) * HID;
    float m = -INFINITY, l = 0.0f;
    float a0 = 0.0f, a1 = 0.0f, a2 = 0.0f, a3 = 0.0f;

    for (int t = 0; t < TCHUNK; ++t) {
        const float* h = hb + (size_t)t * HID;
        if (t + 2 < TCHUNK) __builtin_prefetch(h + 2 * HID + tid * 4, 0, 1);
        float h0 = h[tid], h1 = h[tid + 256], h2 = h[tid + 512], h3 = h[tid + 768];
        // partial dot with alpha
        float p = h0 * sAlpha[tid] + h1 * sAlpha[tid + 256] +
                  h2 * sAlpha[tid + 512] + h3 * sAlpha[tid + 768];
        for (int o = 16; o; o >>= 1) p += __shfl_xor(p, o, 32);
        if (lane == 0) red[wave] = p;
        __syncthreads();
        float s = red[0] + red[1] + red[2] + red[3] + red[4] + red[5] + red[6] + red[7];
        __syncthreads();
        // online softmax update
        float mn = fmaxf(m, s);
        float scale = expf(m - mn);      // 0 on first iteration (m = -inf)
        float w = expf(s - mn);
        l = l * scale + w;
        a0 = a0 * scale + w * h0;
        a1 = a1 * scale + w * h1;
        a2 = a2 * scale + w * h2;
        a3 = a3 * scale + w * h3;
        m = mn;
    }

    float* cp = ctx_part + ((size_t)b * NCHUNK + chunk) * HID;
    cp[tid] = a0; cp[tid + 256] = a1; cp[tid + 512] = a2; cp[tid + 768] = a3;
    if (tid == 0) {
        ml_part[((size_t)b * NCHUNK + chunk) * 2 + 0] = m;
        ml_part[((size_t)b * NCHUNK + chunk) * 2 + 1] = l;
    }
}

// ---------------------------------------------------------------------------
// 5) combine chunk partials (log-sum-exp merge):  ctx[b,:] result
// ---------------------------------------------------------------------------
__global__ void attn_combine_kernel(const float* __restrict__ ctx_part,
                                    const float* __restrict__ ml_part,
                                    float* __restrict__ ctx) {
    const int b = blockIdx.x, tid = threadIdx.x;
    float mj[NCHUNK], lj[NCHUNK];
    float M = -INFINITY;
    for (int j = 0; j < NCHUNK; ++j) {
        mj[j] = ml_part[((size_t)b * NCHUNK + j) * 2 + 0];
        lj[j] = ml_part[((size_t)b * NCHUNK + j) * 2 + 1];
        M = fmaxf(M, mj[j]);
    }
    float wgt[NCHUNK], L = 0.0f;
    for (int j = 0; j < NCHUNK; ++j) {
        wgt[j] = expf(mj[j] - M);
        L += wgt[j] * lj[j];
    }
    float inv = 1.0f / L;
    for (int e = tid; e < HID; e += 256) {
        float v = 0.0f;
        for (int j = 0; j < NCHUNK; ++j)
            v += wgt[j] * ctx_part[((size_t)b * NCHUNK + j) * HID + e];
        ctx[(size_t)b * HID + e] = v * inv;
    }
}

// ---------------------------------------------------------------------------
extern "C" void kernel_launch(void* const* d_in, const int* in_sizes, int n_in,
                              void* d_out, int out_size, void* d_ws, size_t ws_size,
                              hipStream_t stream) {
    (void)in_sizes; (void)n_in; (void)out_size; (void)ws_size;

    const int*   z_t     = (const int*)  d_in[0];
    const float* h_t_1   = (const float*)d_in[1];
    const float* c_t_1   = (const float*)d_in[2];
    const float* h_t_2   = (const float*)d_in[3];
    const float* c_t_2   = (const float*)d_in[4];
    const float* H       = (const float*)d_in[5];
    const float* context = (const float*)d_in[6];
    const float* table   = (const float*)d_in[7];
    const float* W1 = (const float*)d_in[8];
    const float* U1 = (const float*)d_in[9];
    const float* b1 = (const float*)d_in[10];
    const float* W2 = (const float*)d_in[11];
    const float* U2 = (const float*)d_in[12];
    const float* b2 = (const float*)d_in[13];
    const float* Wd1 = (const float*)d_in[14];
    const float* bd1 = (const float*)d_in[15];
    const float* Wd2 = (const float*)d_in[16];
    const float* bd2 = (const float*)d_in[17];
    const float* Wd3 = (const float*)d_in[18];
    const float* bd3 = (const float*)d_in[19];

    // Output layout (return order): out, h1, c1, h2, c2, ctx
    float* out  = (float*)d_out;
    float* h1   = out + (size_t)B_SZ * V_SZ;
    float* c1   = h1 + (size_t)B_SZ * HID;
    float* h2   = c1 + (size_t)B_SZ * HID;
    float* c2   = h2 + (size_t)B_SZ * HID;
    float* ctx2 = c2 + (size_t)B_SZ * HID;

    // Workspace layout (floats)
    float* ws    = (float*)d_ws;
    float* x1    = ws;                             // 256*1536
    float* z1    = x1 + (size_t)B_SZ * 3 * E_SZ;   // 256*4096
    float* z2    = z1 + (size_t)B_SZ * G4;         // 256*4096
    float* alpha = z2 + (size_t)B_SZ * G4;         // 256*1024
    float* ctxp  = alpha + (size_t)B_SZ * HID;     // 256*NCHUNK*1024
    float* mlp   = ctxp + (size_t)B_SZ * NCHUNK * HID;  // 256*NCHUNK*2
    float* ctxr  = mlp + (size_t)B_SZ * NCHUNK * 2;     // 256*1024

    // 1) x1 = [embed | context]
    embed_concat_kernel<<<(B_SZ * 3 * E_SZ) / 256, 256, 0, stream>>>(z_t, context, table, x1);

    // 2) z1 = x1@W1 + h_t_1@U1 + b1
    gemm_f16wmma_kernel<false><<<dim3(G4 / 128, B_SZ / 64), 256, 0, stream>>>(
        x1, W1, 3 * E_SZ, h_t_1, U1, HID, b1, z1, G4, 0);

    // 3) LSTM cell 1 -> h1, c1
    lstm_cell_kernel<<<(B_SZ * HID) / 256, 256, 0, stream>>>(z1, c_t_1, h1, c1);

    // 4) z2 = h1@W2 + h_t_2@U2 + b2
    gemm_f16wmma_kernel<false><<<dim3(G4 / 128, B_SZ / 64), 256, 0, stream>>>(
        h1, W2, HID, h_t_2, U2, HID, b2, z2, G4, 0);

    // 5) LSTM cell 2 -> h2, c2
    lstm_cell_kernel<<<(B_SZ * HID) / 256, 256, 0, stream>>>(z2, c_t_2, h2, c2);

    // 6) alpha = h2@Wd1 + bd1
    gemm_f16wmma_kernel<false><<<dim3(HID / 128, B_SZ / 64), 256, 0, stream>>>(
        h2, Wd1, HID, nullptr, nullptr, 0, bd1, alpha, HID, 0);

    // 7) one-pass attention (single read of H) + combine
    attn_fused_kernel<<<dim3(NCHUNK, B_SZ), 256, 0, stream>>>(H, alpha, ctxp, mlp);
    attn_combine_kernel<<<B_SZ, 256, 0, stream>>>(ctxp, mlp, ctxr);

    // 8) ctx2 = tanh([h2 | ctxr] @ Wd2 + bd2)
    gemm_f16wmma_kernel<false><<<dim3(HID / 128, B_SZ / 64), 256, 0, stream>>>(
        h2, Wd2, HID, ctxr, Wd2 + (size_t)HID * HID, HID, bd2, ctx2, HID, 1);

    // 9) out = ctx2 @ Wd3 + bd3  (V=50257, ragged N; guard only on last tile)
    gemm_f16wmma_kernel<true><<<dim3((V_SZ + 127) / 128, B_SZ / 64), 256, 0, stream>>>(
        ctx2, Wd3, HID, nullptr, nullptr, 0, bd3, out, V_SZ, 0);
}
